// GIN_hybrid_29231547416662
// MI455X (gfx1250) — matrix operand order repaired and checked by reading, not compiled, not run on hardware
//
#include <hip/hip_runtime.h>
#include <hip/hip_bf16.h>
#include <stdint.h>

#define D 128
#define G_NUM 256
#define OUTF 256

typedef __attribute__((ext_vector_type(16))) __bf16 v16bf;
typedef __attribute__((ext_vector_type(8)))  float  v8f;

union Frag { v16bf v; unsigned int u[8]; uint4 q[2]; };

// pack two fp32 -> two bf16 (round-to-nearest, ties away) in one v_perm_b32
__device__ __forceinline__ unsigned int bfpack(float a, float b) {
  unsigned int ua = __float_as_uint(a) + 0x8000u;
  unsigned int ub = __float_as_uint(b) + 0x8000u;
  // result = { ub[31:16], ua[31:16] }
  return __builtin_amdgcn_perm(ub, ua, 0x07060302u);
}
__device__ __forceinline__ unsigned short f2bf(float f) {
  return (unsigned short)((__float_as_uint(f) + 0x8000u) >> 16);
}

// ---------------- weight prep: fp32 [k][n] -> bf16 transposed [n][k] -----------
// wt layout: 6 matrices of 128x128 bf16 (n-major). order: w1_0, w2_0, gw1[0], gw2[0], gw1[1], gw2[1]
__global__ void prep_w_kernel(const float* __restrict__ w1_0, const float* __restrict__ w2_0,
                              const float* __restrict__ gw1, const float* __restrict__ gw2,
                              unsigned short* __restrict__ wt) {
  int t = blockIdx.x * blockDim.x + threadIdx.x;
  if (t >= 6 * D * D) return;
  int mat = t / (D * D);
  int rem = t % (D * D);
  int n = rem / D;      // output col (row of transposed)
  int k = rem % D;      // input dim
  const float* s;
  switch (mat) {
    case 0: s = w1_0;          break;
    case 1: s = w2_0;          break;
    case 2: s = gw1;           break;
    case 3: s = gw2;           break;
    case 4: s = gw1 + D * D;   break;
    default: s = gw2 + D * D;  break;
  }
  wt[t] = f2bf(s[k * D + n]);
}

// ---------------- z = h (vector copy) ------------------------------------------
__global__ void copy_kernel(const float4* __restrict__ in, float4* __restrict__ out, int n4) {
  int i = blockIdx.x * blockDim.x + threadIdx.x;
  if (i < n4) out[i] = in[i];
}

// ---------------- z[dst] += h[src]  (one wave per edge, float4 per lane) --------
__global__ void scatter_add_kernel(const float* __restrict__ h, float* __restrict__ z,
                                   const int* __restrict__ src, const int* __restrict__ dst,
                                   int E) {
  int t = blockIdx.x * blockDim.x + threadIdx.x;
  int e = t >> 5;
  int lane = t & 31;
  if (e >= E) return;
  int s = src[e], d = dst[e];
  const float4 v = ((const float4*)(h + (size_t)s * D))[lane];
  float* zp = z + (size_t)d * D + lane * 4;
  unsafeAtomicAdd(zp + 0, v.x);
  unsafeAtomicAdd(zp + 1, v.y);
  unsafeAtomicAdd(zp + 2, v.z);
  unsafeAtomicAdd(zp + 3, v.w);
}

// ---------------- fused GIN MLP: z <- relu( relu(z@W1) @ W2 )  (in place) -------
// 8 waves / block; wave w handles rows [ (blockIdx*8+w)*16 , +16 ).
// LDS: W1^T bf16 (32KB) + W2^T bf16 (32KB) + per-wave Y staging 16x128 bf16 (8*4KB)
__global__ void __launch_bounds__(256)
gin_mlp_kernel(float* __restrict__ z,
               const unsigned short* __restrict__ w1t,
               const unsigned short* __restrict__ w2t,
               int N) {
  extern __shared__ char smem_raw[];
  unsigned short* sW1 = (unsigned short*)smem_raw;   // D*D
  unsigned short* sW2 = sW1 + D * D;                 // D*D
  unsigned short* sY  = sW2 + D * D;                 // 8 * 16 * D

  const int tid  = threadIdx.x;
  const int wave = tid >> 5;
  const int lane = tid & 31;

  // cooperative load of bf16 transposed weights into LDS (128-bit chunks)
  {
    const uint4* g1 = (const uint4*)w1t;
    const uint4* g2 = (const uint4*)w2t;
    uint4* s1 = (uint4*)sW1;
    uint4* s2 = (uint4*)sW2;
    for (int i = tid; i < D * D / 8; i += 256) { s1[i] = g1[i]; s2[i] = g2[i]; }
  }
  __syncthreads();

  const int rowBlock = blockIdx.x * 8 + wave;
  if (rowBlock * 16 < N) {
    const int row0  = rowBlock * 16;
    const int m     = lane & 15;          // A-layout: row in lanes
    const int kbase = (lane >> 4) * 8;    // A-layout lane-group K offset
    const int kgB   = (lane >> 4) * 16;   // unused; B loads use kbase like A
    const int n_loc = lane & 15;          // B/C-layout: col in lanes

    // ---- A fragments for GEMM1: z rows fp32 -> bf16 packs (ISA A layout) ----
    // per k-step: VGPR j=0..3 hold k = kbase+2j (contig 8 floats), j=4..7 hold +16
    unsigned int aU[4][8];
    const float* zrow = z + (size_t)(row0 + m) * D;
    #pragma unroll
    for (int ks = 0; ks < 4; ++ks) {
      const float4* p0 = (const float4*)(zrow + ks * 32 + kbase);
      const float4* p1 = (const float4*)(zrow + ks * 32 + 16 + kbase);
      float4 fa = p0[0], fb = p0[1], fc = p1[0], fd = p1[1];
      aU[ks][0] = bfpack(fa.x, fa.y);
      aU[ks][1] = bfpack(fa.z, fa.w);
      aU[ks][2] = bfpack(fb.x, fb.y);
      aU[ks][3] = bfpack(fb.z, fb.w);
      aU[ks][4] = bfpack(fc.x, fc.y);
      aU[ks][5] = bfpack(fc.z, fc.w);
      aU[ks][6] = bfpack(fd.x, fd.y);
      aU[ks][7] = bfpack(fd.z, fd.w);
    }

    unsigned short* myY = sY + wave * (16 * D);

    // ---- GEMM1: y = relu(z @ W1), staged to LDS as bf16 row-major [m][n] ----
    #pragma unroll
    for (int nt = 0; nt < 8; ++nt) {
      v8f acc = {};
      const unsigned short* wcol = sW1 + (size_t)(nt * 16 + n_loc) * D;
      #pragma unroll
      for (int ks = 0; ks < 4; ++ks) {
        Frag A, B;
        #pragma unroll
        for (int j = 0; j < 8; ++j) A.u[j] = aU[ks][j];
        B.q[0] = *(const uint4*)(wcol + ks * 32 + kbase);       // k = kbase+0..7
        B.q[1] = *(const uint4*)(wcol + ks * 32 + 16 + kbase);  // k = 16+kbase+0..7
        acc = __builtin_amdgcn_wmma_f32_16x16x32_bf16(false, A.v, false, B.v,
                                                      (short)0, acc, false, false);
      }
      #pragma unroll
      for (int j = 0; j < 8; ++j) {
        float v = acc[j] > 0.f ? acc[j] : 0.f;
        int mo = (lane >> 4) * 8 + j;                // C layout -> row-major restage
        myY[mo * D + nt * 16 + n_loc] = f2bf(v);
      }
    }

    // ---- A fragments for GEMM2 from Y staging (cross-lane transpose via LDS) ----
    unsigned int a2U[4][8];
    #pragma unroll
    for (int ks = 0; ks < 4; ++ks) {
      uint4 qa = *(const uint4*)(myY + m * D + ks * 32 + kbase);        // 8 bf16
      uint4 qb = *(const uint4*)(myY + m * D + ks * 32 + 16 + kbase);   // 8 bf16
      a2U[ks][0] = qa.x; a2U[ks][1] = qa.y; a2U[ks][2] = qa.z; a2U[ks][3] = qa.w;
      a2U[ks][4] = qb.x; a2U[ks][5] = qb.y; a2U[ks][6] = qb.z; a2U[ks][7] = qb.w;
    }

    // ---- GEMM2: h = relu(y @ W2), write back in place ----
    #pragma unroll
    for (int nt = 0; nt < 8; ++nt) {
      v8f acc = {};
      const unsigned short* wcol = sW2 + (size_t)(nt * 16 + n_loc) * D;
      #pragma unroll
      for (int ks = 0; ks < 4; ++ks) {
        Frag A, B;
        #pragma unroll
        for (int j = 0; j < 8; ++j) A.u[j] = a2U[ks][j];
        B.q[0] = *(const uint4*)(wcol + ks * 32 + kbase);
        B.q[1] = *(const uint4*)(wcol + ks * 32 + 16 + kbase);
        acc = __builtin_amdgcn_wmma_f32_16x16x32_bf16(false, A.v, false, B.v,
                                                      (short)0, acc, false, false);
      }
      #pragma unroll
      for (int j = 0; j < 8; ++j) {
        float v = acc[j] > 0.f ? acc[j] : 0.f;
        int mo = (lane >> 4) * 8 + j;
        z[(size_t)(row0 + mo) * D + nt * 16 + n_loc] = v;
      }
    }
  }
}

// ---------------- pooling scratch zero-init ------------------------------------
__global__ void zero_kernel(float* __restrict__ p, int n) {
  int i = blockIdx.x * blockDim.x + threadIdx.x;
  if (i < n) p[i] = 0.0f;
}

// ---------------- per-graph sum / max / count (h >= 0 after relu) ---------------
__global__ void pool_kernel(const float* __restrict__ h, const int* __restrict__ batch,
                            float* __restrict__ gsum, unsigned int* __restrict__ gmaxU,
                            float* __restrict__ cnt, int N) {
  int t = blockIdx.x * blockDim.x + threadIdx.x;
  int node = t >> 5;
  int lane = t & 31;
  if (node >= N) return;
  int g = batch[node];
  const float4 v = ((const float4*)(h + (size_t)node * D))[lane];
  float* sp = gsum + (size_t)g * D + lane * 4;
  unsigned int* mp = gmaxU + (size_t)g * D + lane * 4;
  unsafeAtomicAdd(sp + 0, v.x);
  unsafeAtomicAdd(sp + 1, v.y);
  unsafeAtomicAdd(sp + 2, v.z);
  unsafeAtomicAdd(sp + 3, v.w);
  atomicMax(mp + 0, __float_as_uint(v.x));   // valid order: h >= 0
  atomicMax(mp + 1, __float_as_uint(v.y));
  atomicMax(mp + 2, __float_as_uint(v.z));
  atomicMax(mp + 3, __float_as_uint(v.w));
  if (lane == 0) unsafeAtomicAdd(cnt + g, 1.0f);
}

// ---------------- readout: [gmax | gmean | gsum] @ out_w + out_b ----------------
__global__ void readout_kernel(const float* __restrict__ gsum, const unsigned int* __restrict__ gmaxU,
                               const float* __restrict__ cnt, const float* __restrict__ out_w,
                               const float* __restrict__ out_b, float* __restrict__ out) {
  int t = blockIdx.x * blockDim.x + threadIdx.x;
  if (t >= G_NUM * OUTF) return;
  int g = t / OUTF, o = t % OUTF;
  float c = cnt[g];
  c = c > 1.0f ? c : 1.0f;
  float acc = out_b[o];
  const float* wmax  = out_w;                    // rows [0,128)
  const float* wmean = out_w + (size_t)D * OUTF; // rows [128,256)
  const float* wsum  = out_w + (size_t)2 * D * OUTF;
  for (int k = 0; k < D; ++k) {
    float mx = __uint_as_float(gmaxU[(size_t)g * D + k]);
    float s  = gsum[(size_t)g * D + k];
    acc += mx * wmax[k * OUTF + o] + (s / c) * wmean[k * OUTF + o] + s * wsum[k * OUTF + o];
  }
  out[t] = acc;
}

extern "C" void kernel_launch(void* const* d_in, const int* in_sizes, int n_in,
                              void* d_out, int out_size, void* d_ws, size_t ws_size,
                              hipStream_t stream) {
  const float* x      = (const float*)d_in[0];
  const int*   edge   = (const int*)d_in[1];   // [2, E]
  const int*   batch  = (const int*)d_in[2];
  const float* w1_0   = (const float*)d_in[3];
  const float* w2_0   = (const float*)d_in[4];
  const float* gw1    = (const float*)d_in[5]; // [2,128,128]
  const float* gw2    = (const float*)d_in[6];
  const float* out_w  = (const float*)d_in[7];
  const float* out_b  = (const float*)d_in[8];
  (void)n_in; (void)ws_size;

  const int N = in_sizes[0] / D;      // 50000
  const int E = in_sizes[1] / 2;      // 800000
  const int* src = edge;
  const int* dst = edge + E;

  // ---- workspace carve (~52 MB) ----
  char* ws = (char*)d_ws;
  float* bufA = (float*)ws;                                     // N*D f32
  float* bufB = bufA + (size_t)N * D;                           // N*D f32
  unsigned short* wt = (unsigned short*)(bufB + (size_t)N * D); // 6*D*D bf16
  float* gsum = (float*)(wt + 6 * D * D);                       // G*D
  unsigned int* gmaxU = (unsigned int*)(gsum + G_NUM * D);      // G*D
  float* cnt = (float*)(gmaxU + G_NUM * D);                     // G

  const int TB = 256;
  const size_t SMEM = (size_t)(2 * D * D + 8 * 16 * D) * sizeof(unsigned short); // 96 KB

  // weights -> bf16 transposed
  prep_w_kernel<<<(6 * D * D + TB - 1) / TB, TB, 0, stream>>>(w1_0, w2_0, gw1, gw2, wt);

  const int n4        = N * (D / 4);
  const int copyGrid  = (n4 + TB - 1) / TB;
  const int scatGrid  = ((E * 32) + TB - 1) / TB;
  const int rowBlocks = (N + 15) / 16;
  const int mlpGrid   = (rowBlocks + 7) / 8;

  // ---- layer 0: h_in = x, z = bufA ----
  copy_kernel<<<copyGrid, TB, 0, stream>>>((const float4*)x, (float4*)bufA, n4);
  scatter_add_kernel<<<scatGrid, TB, 0, stream>>>(x, bufA, src, dst, E);
  gin_mlp_kernel<<<mlpGrid, TB, SMEM, stream>>>(bufA, wt + 0 * D * D, wt + 1 * D * D, N);

  // ---- layer 1: h_in = bufA, z = bufB ----
  copy_kernel<<<copyGrid, TB, 0, stream>>>((const float4*)bufA, (float4*)bufB, n4);
  scatter_add_kernel<<<scatGrid, TB, 0, stream>>>(bufA, bufB, src, dst, E);
  gin_mlp_kernel<<<mlpGrid, TB, SMEM, stream>>>(bufB, wt + 2 * D * D, wt + 3 * D * D, N);

  // ---- layer 2: h_in = bufB, z = bufA ----
  copy_kernel<<<copyGrid, TB, 0, stream>>>((const float4*)bufB, (float4*)bufA, n4);
  scatter_add_kernel<<<scatGrid, TB, 0, stream>>>(bufB, bufA, src, dst, E);
  gin_mlp_kernel<<<mlpGrid, TB, SMEM, stream>>>(bufA, wt + 4 * D * D, wt + 5 * D * D, N);

  // ---- pooling + readout (final h = bufA) ----
  const int poolScratch = G_NUM * D * 2 + G_NUM;   // gsum + gmax + cnt
  zero_kernel<<<(poolScratch + TB - 1) / TB, TB, 0, stream>>>(gsum, poolScratch);
  pool_kernel<<<((N * 32) + TB - 1) / TB, TB, 0, stream>>>(bufA, batch, gsum, gmaxU, cnt, N);
  readout_kernel<<<(G_NUM * OUTF + TB - 1) / TB, TB, 0, stream>>>(gsum, gmaxU, cnt, out_w, out_b,
                                                                  (float*)d_out);
}
